// CPEncoder_51719996178824
// MI455X (gfx1250) — matrix operand order repaired and checked by reading, not compiled, not run on hardware
//
#include <hip/hip_runtime.h>
#include <hip/hip_bf16.h>

// ---------------------------------------------------------------------------
// Types for CDNA5 WMMA (wave32): v16bf A/B fragments, v8f accumulator
// ---------------------------------------------------------------------------
typedef __bf16 bf16_t;
typedef __attribute__((ext_vector_type(16))) __bf16 v16bf;
typedef __attribute__((ext_vector_type(8)))  __bf16 v8bf;
typedef __attribute__((ext_vector_type(8)))  float  v8f;

#define B_  256
#define S_  100
#define D_  512
#define H_  8
#define HD_ 64
#define DFF_ 2048
#define L_  4
#define TOK (B_ * S_)          // 25600 rows, divisible by 128

#define LDT 40                 // LDS row stride (bf16 elems), 80B: conflict-staggered
#define TM  128                // workgroup tile M
#define TN  64                 // workgroup tile N

// ---------------------------------------------------------------------------
// Tiled bf16 WMMA GEMM:  C[M,N] = A[M,K] @ W[N,K]^T (+bias) (+relu)
//   A row-major [M,K] bf16, W row-major [N,K] bf16 (weights as stored).
//   Cf (f32) and/or Cb (bf16) outputs, either may be null.
//   Block: 256 threads (8 waves). Workgroup tile 128x64, K-step 32.
//   Each wave owns a 32x32 register tile: 2 A-frags x 2 B-frags -> 4 WMMAs
//   per K-step. LDS double-buffered: one barrier per K-step; global loads for
//   tile k+1 overlap WMMAs on tile k.
// ---------------------------------------------------------------------------
__global__ __launch_bounds__(256)
void k_gemm_bf16(const bf16_t* __restrict__ A, const bf16_t* __restrict__ W,
                 const float* __restrict__ bias,
                 float* __restrict__ Cf, bf16_t* __restrict__ Cb,
                 int M, int N, int K, int relu)
{
    __shared__ bf16_t As[2][TM * LDT];
    __shared__ bf16_t Ws[2][TN * LDT];

    const int tid  = threadIdx.x;
    const int wave = tid >> 5;
    const int lane = tid & 31;
    const int m0 = blockIdx.y * TM;
    const int n0 = blockIdx.x * TN;

    // wave -> 32x32 register tile: rows [mt2*32, +32), cols [ntp*16, +32)
    const int mt2 = wave >> 1;        // 0..3
    const int ntp = (wave & 1) * 2;   // 0 or 2

    const int halfsel = lane >> 4;    // 0/1: which K-half this lane owns
    const int r16     = lane & 15;

    v8f acc00 = {}, acc01 = {}, acc10 = {}, acc11 = {};

    // cooperative loader mapping:
    //   A tile 128x32: each thread owns one row-half (32B): row tid>>1, col (tid&1)*16
    //   W tile  64x32: each thread owns 16B: row tid>>2, col (tid&3)*8
    const int arow = tid >> 1;
    const int acol = (tid & 1) * 16;
    const int wrow = tid >> 2;
    const int wcol = (tid & 3) * 8;

    const bf16_t* Ag = A + (size_t)(m0 + arow) * K + acol;
    const bf16_t* Wg = W + (size_t)(n0 + wrow) * K + wcol;

    // prologue: stage tile 0 into buffer 0
    v8bf ga0 = *(const v8bf*)(Ag);
    v8bf ga1 = *(const v8bf*)(Ag + 8);
    v8bf gw  = *(const v8bf*)(Wg);
    *(v8bf*)(&As[0][arow * LDT + acol])     = ga0;
    *(v8bf*)(&As[0][arow * LDT + acol + 8]) = ga1;
    *(v8bf*)(&Ws[0][wrow * LDT + wcol])     = gw;
    __syncthreads();

    int p = 0;
    for (int k0 = 0; k0 < K; k0 += 32) {
        const bool nxt = (k0 + 32) < K;
        if (nxt) {
            // global loads for the next K-slice (overlap with WMMAs below)
            ga0 = *(const v8bf*)(Ag + k0 + 32);
            ga1 = *(const v8bf*)(Ag + k0 + 40);
            gw  = *(const v8bf*)(Wg + k0 + 32);
            if (k0 + 64 < K) {
                __builtin_prefetch(Ag + k0 + 64, 0, 1);
                __builtin_prefetch(Wg + k0 + 64, 0, 1);
            }
        }

        // --- fragments from buffer p ---
        // A fragment (16x32): lane<16 holds row r16 K{0..7,16..23}, lane>=16 K{8..15,24..31}
        v16bf a0, a1;
        {
            const bf16_t* ap0 = &As[p][((mt2 * 2 + 0) * 16 + r16) * LDT + halfsel * 8];
            const bf16_t* ap1 = &As[p][((mt2 * 2 + 1) * 16 + r16) * LDT + halfsel * 8];
            v8bf lo0 = *(const v8bf*)(ap0);
            v8bf hi0 = *(const v8bf*)(ap0 + 16);
            v8bf lo1 = *(const v8bf*)(ap1);
            v8bf hi1 = *(const v8bf*)(ap1 + 16);
            #pragma unroll
            for (int i = 0; i < 8; ++i) {
                a0[i] = lo0[i]; a0[8 + i] = hi0[i];
                a1[i] = lo1[i]; a1[8 + i] = hi1[i];
            }
        }
        // B fragment (32x16): lane col = r16, lane<16 K{0..7,16..23}, lane>=16 K{8..15,24..31}
        v16bf b0, b1;
        {
            const bf16_t* bp0 = &Ws[p][((ntp + 0) * 16 + r16) * LDT + halfsel * 8];
            const bf16_t* bp1 = &Ws[p][((ntp + 1) * 16 + r16) * LDT + halfsel * 8];
            v8bf lo0 = *(const v8bf*)(bp0);
            v8bf hi0 = *(const v8bf*)(bp0 + 16);
            v8bf lo1 = *(const v8bf*)(bp1);
            v8bf hi1 = *(const v8bf*)(bp1 + 16);
            #pragma unroll
            for (int i = 0; i < 8; ++i) {
                b0[i] = lo0[i]; b0[8 + i] = hi0[i];
                b1[i] = lo1[i]; b1[8 + i] = hi1[i];
            }
        }

        acc00 = __builtin_amdgcn_wmma_f32_16x16x32_bf16(false, a0, false, b0,
                                                        (short)0, acc00, false, false);
        acc01 = __builtin_amdgcn_wmma_f32_16x16x32_bf16(false, a0, false, b1,
                                                        (short)0, acc01, false, false);
        acc10 = __builtin_amdgcn_wmma_f32_16x16x32_bf16(false, a1, false, b0,
                                                        (short)0, acc10, false, false);
        acc11 = __builtin_amdgcn_wmma_f32_16x16x32_bf16(false, a1, false, b1,
                                                        (short)0, acc11, false, false);

        if (nxt) {
            // stage next tile into the other buffer
            *(v8bf*)(&As[p ^ 1][arow * LDT + acol])     = ga0;
            *(v8bf*)(&As[p ^ 1][arow * LDT + acol + 8]) = ga1;
            *(v8bf*)(&Ws[p ^ 1][wrow * LDT + wcol])     = gw;
        }
        __syncthreads();
        p ^= 1;
    }

    // Epilogue: C/D layout -> lane col = r16, rows = halfsel*8 + r
    const int col     = r16;
    const int rowbase = halfsel * 8;
    #pragma unroll
    for (int j = 0; j < 2; ++j) {
        const int n = n0 + (ntp + j) * 16 + col;
        const float bv = bias ? bias[n] : 0.0f;
        const v8f accA = j ? acc01 : acc00;   // row-subtile 0
        const v8f accB = j ? acc11 : acc10;   // row-subtile 1
        #pragma unroll
        for (int r = 0; r < 8; ++r) {
            const int mA = m0 + (mt2 * 2 + 0) * 16 + rowbase + r;
            const int mB = m0 + (mt2 * 2 + 1) * 16 + rowbase + r;
            float v0 = accA[r] + bv;
            float v1 = accB[r] + bv;
            if (relu) { v0 = fmaxf(v0, 0.0f); v1 = fmaxf(v1, 0.0f); }
            const size_t iA = (size_t)mA * N + n;
            const size_t iB = (size_t)mB * N + n;
            if (Cf) { Cf[iA] = v0; Cf[iB] = v1; }
            if (Cb) { Cb[iA] = (bf16_t)v0; Cb[iB] = (bf16_t)v1; }
        }
    }
}

// ---------------------------------------------------------------------------
// Positional encoding + embedding add; produce f32 and bf16 activations
// ---------------------------------------------------------------------------
__global__ void k_pe_add(const float* __restrict__ emb,
                         float* __restrict__ xo, bf16_t* __restrict__ xob, int total)
{
    int idx = blockIdx.x * blockDim.x + threadIdx.x;
    if (idx >= total) return;
    int c = idx & (D_ - 1);
    int s = (idx >> 9) % S_;
    int i2 = c & ~1;
    float arg = (float)s * ((float)i2 * (-4.605170185988091f / (float)D_)); // -log(100)/D
    float pe  = (c & 1) ? __cosf(arg) : __sinf(arg);
    float v = emb[idx] + pe;
    xo[idx]  = v;
    xob[idx] = (bf16_t)v;
}

__global__ void k_cvt_bf16(const float* __restrict__ src, bf16_t* __restrict__ dst, int n)
{
    int idx = blockIdx.x * blockDim.x + threadIdx.x;
    if (idx < n) dst[idx] = (bf16_t)src[idx];
}

// ---------------------------------------------------------------------------
// Attention: one block per (b,h). K/V staged in LDS; flash-style online
// softmax per query in f32; writes bf16 [B*S, D] (head-interleaved) for the
// out-projection WMMA GEMM.
// ---------------------------------------------------------------------------
__global__ __launch_bounds__(128)
void k_attention(const float* __restrict__ qkv, const int* __restrict__ inputs,
                 bf16_t* __restrict__ attn_out)
{
    __shared__ float Ks[S_ * HD_];
    __shared__ float Vs[S_ * HD_];
    __shared__ float padf[S_];

    const int bh = blockIdx.x;
    const int b  = bh >> 3;       // H_ = 8
    const int h  = bh & 7;
    const int tid = threadIdx.x;
    const size_t base = (size_t)b * S_ * (3 * D_) + (size_t)h * HD_;

    for (int idx = tid; idx < S_ * HD_; idx += 128) {
        int s = idx >> 6, d = idx & 63;
        Ks[idx] = qkv[base + (size_t)s * (3 * D_) + D_     + d];
        Vs[idx] = qkv[base + (size_t)s * (3 * D_) + 2 * D_ + d];
    }
    for (int s = tid; s < S_; s += 128)
        padf[s] = (inputs[b * S_ + s] == 0) ? 1.0f : 0.0f;
    __syncthreads();

    if (tid < S_) {
        const int q = tid;
        float qv[HD_];
        #pragma unroll
        for (int d = 0; d < HD_; ++d)
            qv[d] = qkv[base + (size_t)q * (3 * D_) + d];

        float m = -1e30f, l = 0.0f;
        float acc[HD_];
        #pragma unroll
        for (int d = 0; d < HD_; ++d) acc[d] = 0.0f;

        for (int k = 0; k < S_; ++k) {
            if (padf[k] != 0.0f) continue;
            float s = 0.0f;
            #pragma unroll
            for (int d = 0; d < HD_; ++d) s += qv[d] * Ks[k * HD_ + d];
            s *= 0.125f;                       // 1/sqrt(HD)
            float nm = fmaxf(m, s);
            float sc = __expf(m - nm);
            float p  = __expf(s - nm);
            l = l * sc + p;
            #pragma unroll
            for (int d = 0; d < HD_; ++d)
                acc[d] = acc[d] * sc + p * Vs[k * HD_ + d];
            m = nm;
        }
        const float inv = 1.0f / l;
        bf16_t* op = attn_out + ((size_t)b * S_ + q) * D_ + h * HD_;
        #pragma unroll
        for (int d = 0; d < HD_; ++d) op[d] = (bf16_t)(acc[d] * inv);
    }
}

// ---------------------------------------------------------------------------
// Residual + LayerNorm: x = LN(x + y); write f32 and bf16. One block per row.
// ---------------------------------------------------------------------------
__global__ __launch_bounds__(256)
void k_add_ln(const float* __restrict__ x, const float* __restrict__ y,
              const float* __restrict__ g, const float* __restrict__ bta,
              float* __restrict__ xo, bf16_t* __restrict__ xob)
{
    __shared__ float red[256];
    const int row = blockIdx.x;
    const int tid = threadIdx.x;
    const size_t base = (size_t)row * D_;

    float v0 = x[base + tid]       + y[base + tid];
    float v1 = x[base + 256 + tid] + y[base + 256 + tid];

    red[tid] = v0 + v1;
    __syncthreads();
    for (int s = 128; s > 0; s >>= 1) {
        if (tid < s) red[tid] += red[tid + s];
        __syncthreads();
    }
    const float mu = red[0] * (1.0f / (float)D_);
    __syncthreads();

    const float d0 = v0 - mu, d1 = v1 - mu;
    red[tid] = d0 * d0 + d1 * d1;
    __syncthreads();
    for (int s = 128; s > 0; s >>= 1) {
        if (tid < s) red[tid] += red[tid + s];
        __syncthreads();
    }
    const float rs = rsqrtf(red[0] * (1.0f / (float)D_) + 1e-5f);

    const float o0 = d0 * rs * g[tid]       + bta[tid];
    const float o1 = d1 * rs * g[256 + tid] + bta[256 + tid];
    xo[base + tid]        = o0;
    xo[base + 256 + tid]  = o1;
    xob[base + tid]       = (bf16_t)o0;
    xob[base + 256 + tid] = (bf16_t)o1;
}

// ---------------------------------------------------------------------------
// Final pooling: (x @ W^T).sum(-1) == x . colsum(W). Precompute column sums.
// ---------------------------------------------------------------------------
__global__ void k_colsum(const float* __restrict__ attn_w, const float* __restrict__ attn_b,
                         float* __restrict__ w1s, float* __restrict__ w2s,
                         float* __restrict__ bsum)
{
    int d = blockIdx.x * 256 + threadIdx.x;
    if (d < D_) {
        float s1 = 0.0f, s2 = 0.0f;
        for (int j = 0; j < D_; ++j) {
            s1 += attn_w[(size_t)j * (2 * D_) + d];
            s2 += attn_w[(size_t)j * (2 * D_) + D_ + d];
        }
        w1s[d] = s1;
        w2s[d] = s2;
    }
    if (blockIdx.x == 0 && threadIdx.x == 0) {
        float s = 0.0f;
        for (int j = 0; j < D_; ++j) s += attn_b[j];
        *bsum = s;
    }
}

__global__ __launch_bounds__(128)
void k_pool(const float* __restrict__ x, const int* __restrict__ inputs,
            const float* __restrict__ w1s, const float* __restrict__ w2s,
            const float* __restrict__ bsum, float* __restrict__ out)
{
    __shared__ float sc[S_];
    __shared__ float p[S_];
    __shared__ float sq_sh;
    const int b = blockIdx.x;
    const int tid = threadIdx.x;
    const size_t xb = (size_t)b * S_ * D_;

    if (tid < S_) {
        float s = 0.0f;
        const float* xr = x + xb + (size_t)tid * D_;
        for (int d = 0; d < D_; ++d) s += xr[d] * w2s[d];
        sc[tid] = s;
    }
    if (tid == 0) {
        float s = 0.0f;
        const float* xq = x + xb + (size_t)(S_ - 1) * D_;
        for (int d = 0; d < D_; ++d) s += xq[d] * w1s[d];
        sq_sh = s + *bsum;
    }
    __syncthreads();

    if (tid == 0) {
        float m = -1e30f;
        for (int s = 0; s < S_; ++s)
            if (inputs[b * S_ + s] != 0) m = fmaxf(m, sq_sh + sc[s]);
        float l = 0.0f;
        for (int s = 0; s < S_; ++s) {
            float pv = (inputs[b * S_ + s] != 0) ? __expf(sq_sh + sc[s] - m) : 0.0f;
            p[s] = pv; l += pv;
        }
        const float inv = 1.0f / l;
        for (int s = 0; s < S_; ++s) p[s] *= inv;
    }
    __syncthreads();

    for (int d = tid; d < D_; d += 128) {
        float a = 0.0f;
        for (int s = 0; s < S_; ++s) a += p[s] * x[xb + (size_t)s * D_ + d];
        out[(size_t)b * D_ + d] = a;
    }
}

// ---------------------------------------------------------------------------
// Host launcher
// ---------------------------------------------------------------------------
extern "C" void kernel_launch(void* const* d_in, const int* in_sizes, int n_in,
                              void* d_out, int out_size, void* d_ws, size_t ws_size,
                              hipStream_t stream)
{
    (void)in_sizes; (void)n_in; (void)out_size; (void)ws_size;

    const int*   inputs = (const int*)  d_in[0];
    const float* emb    = (const float*)d_in[1];
    const float* in_w   = (const float*)d_in[2];
    const float* in_b   = (const float*)d_in[3];
    const float* outp_w = (const float*)d_in[4];
    const float* outp_b = (const float*)d_in[5];
    const float* l1_w   = (const float*)d_in[6];
    const float* l1_b   = (const float*)d_in[7];
    const float* l2_w   = (const float*)d_in[8];
    const float* l2_b   = (const float*)d_in[9];
    const float* ln1w   = (const float*)d_in[10];
    const float* ln1b   = (const float*)d_in[11];
    const float* ln2w   = (const float*)d_in[12];
    const float* ln2b   = (const float*)d_in[13];
    const float* attn_w = (const float*)d_in[14];
    const float* attn_b = (const float*)d_in[15];
    float* out = (float*)d_out;

    char* ws = (char*)d_ws;
    size_t off = 0;
    auto alloc = [&](size_t bytes) -> void* {
        void* p = ws + off;
        off = (off + bytes + 255) & ~(size_t)255;
        return p;
    };

    float*  xf    = (float*) alloc((size_t)TOK * D_ * 4);
    bf16_t* xb16  = (bf16_t*)alloc((size_t)TOK * D_ * 2);
    float*  qkv   = (float*) alloc((size_t)TOK * 3 * D_ * 4);
    bf16_t* attnb = (bf16_t*)alloc((size_t)TOK * D_ * 2);
    float*  yf    = (float*) alloc((size_t)TOK * D_ * 4);
    bf16_t* h1b   = (bf16_t*)alloc((size_t)TOK * DFF_ * 2);
    bf16_t* winb  = (bf16_t*)alloc((size_t)L_ * 3 * D_ * D_ * 2);
    bf16_t* woutb = (bf16_t*)alloc((size_t)L_ * D_ * D_ * 2);
    bf16_t* wl1b  = (bf16_t*)alloc((size_t)L_ * DFF_ * D_ * 2);
    bf16_t* wl2b  = (bf16_t*)alloc((size_t)L_ * D_ * DFF_ * 2);
    float*  w1s   = (float*) alloc(D_ * 4);
    float*  w2s   = (float*) alloc(D_ * 4);
    float*  bsum  = (float*) alloc(4);

    // weight conversion f32 -> bf16 (all layers at once)
    {
        int n;
        n = L_ * 3 * D_ * D_; k_cvt_bf16<<<(n + 255) / 256, 256, 0, stream>>>(in_w,   winb,  n);
        n = L_ * D_ * D_;     k_cvt_bf16<<<(n + 255) / 256, 256, 0, stream>>>(outp_w, woutb, n);
        n = L_ * DFF_ * D_;   k_cvt_bf16<<<(n + 255) / 256, 256, 0, stream>>>(l1_w,   wl1b,  n);
        n = L_ * D_ * DFF_;   k_cvt_bf16<<<(n + 255) / 256, 256, 0, stream>>>(l2_w,   wl2b,  n);
    }

    // x = emb + PE
    {
        int n = TOK * D_;
        k_pe_add<<<(n + 255) / 256, 256, 0, stream>>>(emb, xf, xb16, n);
    }

    const dim3 blk(256);
    for (int i = 0; i < L_; ++i) {
        // QKV projection: [TOK,512] x [1536,512]^T
        dim3 g1(3 * D_ / TN, TOK / TM);
        k_gemm_bf16<<<g1, blk, 0, stream>>>(xb16, winb + (size_t)i * 3 * D_ * D_,
                                            in_b + (size_t)i * 3 * D_,
                                            qkv, nullptr, TOK, 3 * D_, D_, 0);
        // attention per (b,h)
        k_attention<<<B_ * H_, 128, 0, stream>>>(qkv, inputs, attnb);
        // out projection
        dim3 g2(D_ / TN, TOK / TM);
        k_gemm_bf16<<<g2, blk, 0, stream>>>(attnb, woutb + (size_t)i * D_ * D_,
                                            outp_b + (size_t)i * D_,
                                            yf, nullptr, TOK, D_, D_, 0);
        // x = LN(x + attn)
        k_add_ln<<<TOK, 256, 0, stream>>>(xf, yf, ln1w + (size_t)i * D_, ln1b + (size_t)i * D_,
                                          xf, xb16);
        // FFN up + ReLU (bf16 out only)
        dim3 g3(DFF_ / TN, TOK / TM);
        k_gemm_bf16<<<g3, blk, 0, stream>>>(xb16, wl1b + (size_t)i * DFF_ * D_,
                                            l1_b + (size_t)i * DFF_,
                                            nullptr, h1b, TOK, DFF_, D_, 1);
        // FFN down
        dim3 g4(D_ / TN, TOK / TM);
        k_gemm_bf16<<<g4, blk, 0, stream>>>(h1b, wl2b + (size_t)i * D_ * DFF_,
                                            l2_b + (size_t)i * D_,
                                            yf, nullptr, TOK, D_, DFF_, 0);
        // x = LN(x + ff)
        k_add_ln<<<TOK, 256, 0, stream>>>(xf, yf, ln2w + (size_t)i * D_, ln2b + (size_t)i * D_,
                                          xf, xb16);
    }

    // final attention pooling
    k_colsum<<<2, 256, 0, stream>>>(attn_w, attn_b, w1s, w2s, bsum);
    k_pool<<<B_, 128, 0, stream>>>(xf, inputs, w1s, w2s, bsum, out);
}
